// VectorQuantizer_91147795955716
// MI455X (gfx1250) — matrix operand order repaired and checked by reading, not compiled, not run on hardware
//
#include <hip/hip_runtime.h>

// -------------------- problem constants --------------------
#define DIM        256        // e_dim
#define NE         1024       // n_embed
#define NROWS      65536      // 16*4096 flattened tokens
#define Z_ELEMS    16777216   // NROWS * DIM
#define ROWS_PER_BLOCK 64     // 2 waves * 32 rows per wave
#define LDS_STRIDE 264        // 256 + 8 halves pad (keeps 16B alignment, spreads banks)

typedef __attribute__((ext_vector_type(16))) __bf16 bf16x16;
typedef __attribute__((ext_vector_type(8)))  float  f32x8;

union Frag { uint4 q[2]; bf16x16 v; };

__device__ __forceinline__ unsigned short f32_to_bf16(float f) {
    unsigned int u = __float_as_uint(f);
    // round-to-nearest-even truncation to bf16
    u += 0x7FFFu + ((u >> 16) & 1u);
    return (unsigned short)(u >> 16);
}

// -------------------- kernel 1: codebook prep --------------------
// Per block: one codebook row. Convert f32 -> bf16 into scratch, compute ||e||^2.
__global__ void __launch_bounds__(256) vq_prep_kernel(const float* __restrict__ emb,
                                                      unsigned short* __restrict__ emb16,
                                                      float* __restrict__ enorm) {
    __shared__ float red[256];
    const int tid = threadIdx.x;
    const int row = blockIdx.x;
    float v = emb[row * DIM + tid];
    emb16[row * DIM + tid] = f32_to_bf16(v);
    red[tid] = v * v;
    __syncthreads();
    for (int o = 128; o > 0; o >>= 1) {
        if (tid < o) red[tid] += red[tid + o];
        __syncthreads();
    }
    if (tid == 0) enorm[row] = red[0];
}

// -------------------- kernel 2: WMMA distance + argmax --------------------
// Block = 64 threads (2 waves). Block stages a 64x256 bf16 z-tile in LDS; each
// wave owns TWO 16-row M-tiles (32 rows) so every B fragment is reused by two
// independent WMMA chains (halves L2 codebook traffic, hides B-load latency).
// Per n-tile of 16 codebook rows: K-loop of 8 x (2 x v_wmma_f32_16x16x32_bf16),
// then running argmax of (||e||^2 - 2 z.e) (||z||^2 is row-constant -> drops).
__global__ void __launch_bounds__(64) vq_argmax_kernel(const float* __restrict__ z,
                                                       const unsigned short* __restrict__ emb16,
                                                       const float* __restrict__ enorm,
                                                       int* __restrict__ idx_ws,
                                                       float* __restrict__ idx_out) {
    __shared__ unsigned short zt[ROWS_PER_BLOCK * LDS_STRIDE];

    const int tid  = threadIdx.x;
    const int lane = tid & 31;
    const int w    = tid >> 5;            // wave 0/1
    const long zbase = (long)blockIdx.x * ROWS_PER_BLOCK * DIM;

    // Stage 64x256 f32 -> bf16 into LDS (coalesced global reads).
    for (int i = tid; i < ROWS_PER_BLOCK * DIM; i += 64) {
        const int r = i >> 8;
        const int c = i & (DIM - 1);
        zt[r * LDS_STRIDE + c] = f32_to_bf16(z[zbase + i]);
    }
    __syncthreads();

    const int m15 = lane & 15;            // row/col within 16-wide tile
    const int hi  = lane >> 4;            // lane group 0 / 1
    const unsigned short* arow0 = &zt[(w * 32 + m15) * LDS_STRIDE];
    const unsigned short* arow1 = arow0 + 16 * LDS_STRIDE;

    float bestVal0[8], bestVal1[8];
    int   bestIdx0[8], bestIdx1[8];
#pragma unroll
    for (int v = 0; v < 8; ++v) {
        bestVal0[v] = -3.402823466e38f; bestIdx0[v] = 0;
        bestVal1[v] = -3.402823466e38f; bestIdx1[v] = 0;
    }

    for (int n0 = 0; n0 < NE; n0 += 16) {
        f32x8 acc0 = {0.f, 0.f, 0.f, 0.f, 0.f, 0.f, 0.f, 0.f};
        f32x8 acc1 = {0.f, 0.f, 0.f, 0.f, 0.f, 0.f, 0.f, 0.f};
        const unsigned short* brow = emb16 + (n0 + m15) * DIM;
#pragma unroll
        for (int k0 = 0; k0 < DIM; k0 += 32) {
            Frag a0, a1, b;
            // A fragments (16x32 bf16, ISA layout): lanes<16 take K k0..+7 /
            // k0+16..+23, lanes>=16 take k0+8..+15 / k0+24..+31. 16B LDS loads.
            const int ak = k0 + hi * 8;
            a0.q[0] = *(const uint4*)(arow0 + ak);
            a0.q[1] = *(const uint4*)(arow0 + ak + 16);
            a1.q[0] = *(const uint4*)(arow1 + ak);
            a1.q[1] = *(const uint4*)(arow1 + ak + 16);
            // B fragment (32x16 bf16): lane c holds column n0+(c&15), same K
            // split as A: k0+8*hi..+7 / k0+16+8*hi..+7 -> row-major read.
            const int bk = k0 + hi * 8;
            const uint4* bp = (const uint4*)(brow + bk);
            b.q[0] = bp[0];
            b.q[1] = bp[2];
            // Two independent accumulator chains sharing one B fragment.
            acc0 = __builtin_amdgcn_wmma_f32_16x16x32_bf16(false, a0.v, false, b.v,
                                                           (short)0, acc0, false, false);
            acc1 = __builtin_amdgcn_wmma_f32_16x16x32_bf16(false, a1.v, false, b.v,
                                                           (short)0, acc1, false, false);
        }
        const float en = enorm[n0 + m15];
        const int   n  = n0 + m15;
#pragma unroll
        for (int v = 0; v < 8; ++v) {
            const float s0 = en - 2.0f * acc0[v];
            const float s1 = en - 2.0f * acc1[v];
            if (s0 > bestVal0[v]) { bestVal0[v] = s0; bestIdx0[v] = n; }  // '>' keeps first max
            if (s1 > bestVal1[v]) { bestVal1[v] = s1; bestIdx1[v] = n; }
        }
    }

    // Cross-lane argmax within each 16-lane group (rows v for lanes 0-15,
    // rows v+8 for lanes 16-31), min-index tie-break (jnp.argmax semantics).
#pragma unroll
    for (int v = 0; v < 8; ++v) {
        float val0 = bestVal0[v]; int idx0 = bestIdx0[v];
        float val1 = bestVal1[v]; int idx1 = bestIdx1[v];
#pragma unroll
        for (int off = 8; off >= 1; off >>= 1) {
            const float ov0 = __shfl_xor(val0, off, 16);
            const int   oi0 = __shfl_xor(idx0, off, 16);
            const float ov1 = __shfl_xor(val1, off, 16);
            const int   oi1 = __shfl_xor(idx1, off, 16);
            if (ov0 > val0 || (ov0 == val0 && oi0 < idx0)) { val0 = ov0; idx0 = oi0; }
            if (ov1 > val1 || (ov1 == val1 && oi1 < idx1)) { val1 = ov1; idx1 = oi1; }
        }
        bestIdx0[v] = idx0;
        bestIdx1[v] = idx1;
    }

    if (m15 == 0) {
        const int mbase0 = blockIdx.x * ROWS_PER_BLOCK + w * 32 + hi * 8;
        const int mbase1 = mbase0 + 16;
#pragma unroll
        for (int v = 0; v < 8; ++v) {
            idx_ws[mbase0 + v]  = bestIdx0[v];
            idx_out[mbase0 + v] = (float)bestIdx0[v];
            idx_ws[mbase1 + v]  = bestIdx1[v];
            idx_out[mbase1 + v] = (float)bestIdx1[v];
        }
    }
}

// -------------------- kernel 3: gather z_q + partial SSE --------------------
// 16 floats (4 x float4) per thread; each thread's span stays within one token row.
__global__ void __launch_bounds__(256) vq_gather_kernel(const float* __restrict__ z,
                                                        const float* __restrict__ emb,
                                                        const int* __restrict__ idx_ws,
                                                        float* __restrict__ zq,
                                                        float* __restrict__ partials) {
    __shared__ float red[256];
    const int tid   = threadIdx.x;
    const long base = ((long)blockIdx.x * 256 + tid) * 16;
    const int row   = (int)(base >> 8);
    const int col   = (int)(base & (DIM - 1));
    const int idx   = idx_ws[row];

    const float4* zp = (const float4*)(z + base);
    const float4* ep = (const float4*)(emb + (long)idx * DIM + col);
    float4*       qp = (float4*)(zq + base);

    float s = 0.f;
#pragma unroll
    for (int i = 0; i < 4; ++i) {
        const float4 e4 = ep[i];
        const float4 z4 = zp[i];
        qp[i] = e4;
        const float dx = e4.x - z4.x, dy = e4.y - z4.y;
        const float dz = e4.z - z4.z, dw = e4.w - z4.w;
        s += dx * dx + dy * dy + dz * dz + dw * dw;
    }
    red[tid] = s;
    __syncthreads();
    for (int o = 128; o > 0; o >>= 1) {
        if (tid < o) red[tid] += red[tid + o];
        __syncthreads();
    }
    if (tid == 0) partials[blockIdx.x] = red[0];
}

// -------------------- kernel 4: deterministic final loss reduction --------------------
__global__ void __launch_bounds__(256) vq_loss_kernel(const float* __restrict__ partials,
                                                      float* __restrict__ losses) {
    __shared__ float red[256];
    const int tid = threadIdx.x;
    float s = 0.f;
    for (int i = tid; i < 4096; i += 256) s += partials[i];
    red[tid] = s;
    __syncthreads();
    for (int o = 128; o > 0; o >>= 1) {
        if (tid < o) red[tid] += red[tid + o];
        __syncthreads();
    }
    if (tid == 0) {
        const float mean = red[0] * (1.0f / (float)Z_ELEMS);
        losses[0] = mean;          // vq_loss
        losses[1] = 0.25f * mean;  // commitment_loss (BETA = 0.25)
    }
}

// -------------------- launch --------------------
extern "C" void kernel_launch(void* const* d_in, const int* in_sizes, int n_in,
                              void* d_out, int out_size, void* d_ws, size_t ws_size,
                              hipStream_t stream) {
    (void)in_sizes; (void)n_in; (void)out_size; (void)ws_size;
    const float* z   = (const float*)d_in[0];  // [16,4096,256] f32
    const float* emb = (const float*)d_in[1];  // [1024,256] f32

    float* out    = (float*)d_out;
    float* zq     = out;                 // Z_ELEMS floats
    float* losses = out + Z_ELEMS;       // 2 floats
    float* idxf   = out + Z_ELEMS + 2;   // NROWS floats (indices as f32)

    // scratch layout (all 16B aligned)
    char* ws = (char*)d_ws;
    unsigned short* emb16   = (unsigned short*)ws;            // 1024*256*2 = 524288 B
    float*          enorm   = (float*)(ws + 524288);          // 4096 B
    int*            idx_ws  = (int*)(ws + 528384);            // 65536*4 = 262144 B
    float*          partials= (float*)(ws + 790528);          // 4096*4 = 16384 B

    vq_prep_kernel  <<<NE,   256, 0, stream>>>(emb, emb16, enorm);
    vq_argmax_kernel<<<NROWS / ROWS_PER_BLOCK, 64, 0, stream>>>(z, emb16, enorm, idx_ws, idxf);
    vq_gather_kernel<<<Z_ELEMS / (256 * 16), 256, 0, stream>>>(z, emb, idx_ws, zq, partials);
    vq_loss_kernel  <<<1,    256, 0, stream>>>(partials, losses);
}
